// BitLinear_11570641895430
// MI455X (gfx1250) — 8-bit matrix operand order repaired and checked by reading, compiled, not run on hardware
//
#include <hip/hip_runtime.h>
#include <stdint.h>


typedef int  v8i __attribute__((ext_vector_type(8)));
typedef int  v4i __attribute__((ext_vector_type(4)));

#define FAN_IN    4096
#define FAN_OUT   16384
#define NTOK      8192
#define KTILES    (FAN_IN / 64)
#define NT_TILES  (NTOK / 16)
#define MT_TILES  (FAN_OUT / 16)

#define NPART     2048
#define OFF_PART  0
#define OFF_ALPHA (8 * 1024)
#define OFF_GAMMA (16 * 1024)
#define OFF_QX    (64 * 1024)
#define QX_BYTES  ((size_t)NT_TILES * KTILES * 1024)
#define OFF_QW    (OFF_QX + QX_BYTES)

__global__ void absmean_partial(const float* __restrict__ w,
                                float* __restrict__ part) {
    __shared__ float red[256];
    const float4* w4 = (const float4*)w;
    const int nthreads = NPART * 256;
    int t = blockIdx.x * 256 + threadIdx.x;
    float s = 0.f;
#pragma unroll 4
    for (int i = 0; i < 32; ++i) {
        float4 v = w4[t + i * nthreads];
        s += fabsf(v.x) + fabsf(v.y) + fabsf(v.z) + fabsf(v.w);
    }
    red[threadIdx.x] = s;
    __syncthreads();
    for (int off = 128; off > 0; off >>= 1) {
        if (threadIdx.x < off) red[threadIdx.x] += red[threadIdx.x + off];
        __syncthreads();
    }
    if (threadIdx.x == 0) part[blockIdx.x] = red[0];
}

__global__ void absmean_final(const float* __restrict__ part,
                              float* __restrict__ alpha) {
    __shared__ float red[256];
    float s = 0.f;
    for (int i = threadIdx.x; i < NPART; i += 256) s += part[i];
    red[threadIdx.x] = s;
    __syncthreads();
    for (int off = 128; off > 0; off >>= 1) {
        if (threadIdx.x < off) red[threadIdx.x] += red[threadIdx.x + off];
        __syncthreads();
    }
    if (threadIdx.x == 0)
        alpha[0] = red[0] / (float)((long)FAN_OUT * FAN_IN) + 1e-8f;
}

__global__ void rowmax(const float* __restrict__ x, float* __restrict__ gamma) {
    __shared__ float red[256];
    const float4* row = (const float4*)(x + (size_t)blockIdx.x * FAN_IN);
    float m = 0.f;
    for (int i = threadIdx.x; i < FAN_IN / 4; i += 256) {
        float4 v = row[i];
        m = fmaxf(m, fmaxf(fmaxf(fabsf(v.x), fabsf(v.y)),
                           fmaxf(fabsf(v.z), fabsf(v.w))));
    }
    red[threadIdx.x] = m;
    __syncthreads();
    for (int off = 128; off > 0; off >>= 1) {
        if (threadIdx.x < off)
            red[threadIdx.x] = fmaxf(red[threadIdx.x], red[threadIdx.x + off]);
        __syncthreads();
    }
    if (threadIdx.x == 0) gamma[blockIdx.x] = fmaxf(red[0], 1e-8f);
}

__global__ void quant_x(const float* __restrict__ x,
                        const float* __restrict__ gamma,
                        signed char* __restrict__ qx) {
    int t = blockIdx.x * 256 + threadIdx.x;
    int frag = t >> 5;
    int lane = t & 31;
    int nt = frag >> 6;
    int kt = frag & (KTILES - 1);
    int row = nt * 16 + (lane & 15);
    float inv = 127.0f / gamma[row];
    int kb = kt * 64 + ((lane >> 4) << 3);
    const float* xr = x + (size_t)row * FAN_IN;
    int dw[8];
#pragma unroll
    for (int v = 0; v < 8; ++v) {
        int kk = kb + ((v >> 1) << 4) + ((v & 1) << 2);
        float4 f = *(const float4*)(xr + kk);
        int b0 = (int)__builtin_rintf(f.x * inv);
        int b1 = (int)__builtin_rintf(f.y * inv);
        int b2 = (int)__builtin_rintf(f.z * inv);
        int b3 = (int)__builtin_rintf(f.w * inv);
        dw[v] = (b0 & 255) | ((b1 & 255) << 8) | ((b2 & 255) << 16) | (b3 << 24);
    }
    v4i* dst = (v4i*)(qx + (size_t)frag * 1024 + lane * 32);
    dst[0] = (v4i){dw[0], dw[1], dw[2], dw[3]};
    dst[1] = (v4i){dw[4], dw[5], dw[6], dw[7]};
}

__global__ void quant_w(const float* __restrict__ w,
                        const float* __restrict__ alpha,
                        signed char* __restrict__ qw) {
    int t = blockIdx.x * 256 + threadIdx.x;
    int frag = t >> 5;
    int lane = t & 31;
    int mt = frag >> 6;
    int kt = frag & (KTILES - 1);
    int m = mt * 16 + (lane & 15);
    float invA = 1.0f / alpha[0];
    int kb = kt * 64 + ((lane >> 4) << 3);
    const float* wr = w + (size_t)m * FAN_IN;
    int dw[8];
#pragma unroll
    for (int v = 0; v < 8; ++v) {
        int kk = kb + ((v >> 1) << 4) + ((v & 1) << 2);
        float4 f = *(const float4*)(wr + kk);
        int b0 = min(1, max(-1, (int)__builtin_rintf(f.x * invA)));
        int b1 = min(1, max(-1, (int)__builtin_rintf(f.y * invA)));
        int b2 = min(1, max(-1, (int)__builtin_rintf(f.z * invA)));
        int b3 = min(1, max(-1, (int)__builtin_rintf(f.w * invA)));
        dw[v] = (b0 & 255) | ((b1 & 255) << 8) | ((b2 & 255) << 16) | (b3 << 24);
    }
    v4i* dst = (v4i*)(qw + (size_t)frag * 1024 + lane * 32);
    dst[0] = (v4i){dw[0], dw[1], dw[2], dw[3]};
    dst[1] = (v4i){dw[4], dw[5], dw[6], dw[7]};
}

__global__ __launch_bounds__(256)
void gemm_iu8(const signed char* __restrict__ qx,
              const signed char* __restrict__ qw,
              const float* __restrict__ gamma,
              const float* __restrict__ alpha,
              float* __restrict__ out) {
    const int lane = threadIdx.x & 31;
    const int wave = threadIdx.x >> 5;
    const int wn = wave >> 2;
    const int wm = wave & 3;
    const int ntBase = blockIdx.y * 8 + wn * 4;
    const int mtBase = blockIdx.x * 16 + wm * 4;

    const v8i zero = (v8i){0, 0, 0, 0, 0, 0, 0, 0};
    v8i acc[4][4];
#pragma unroll
    for (int i = 0; i < 4; ++i)
#pragma unroll
        for (int j = 0; j < 4; ++j) acc[i][j] = zero;

    const size_t laneOff = (size_t)lane * 32;
    for (int kt = 0; kt < KTILES; ++kt) {
        v8i A[4], B[4];
#pragma unroll
        for (int i = 0; i < 4; ++i)
            A[i] = *(const v8i*)(qx + ((size_t)(ntBase + i) * KTILES + kt) * 1024 + laneOff);
#pragma unroll
        for (int j = 0; j < 4; ++j)
            B[j] = *(const v8i*)(qw + ((size_t)(mtBase + j) * KTILES + kt) * 1024 + laneOff);
#pragma unroll
        for (int i = 0; i < 4; ++i)
#pragma unroll
            for (int j = 0; j < 4; ++j)
                acc[i][j] = __builtin_amdgcn_wmma_i32_16x16x64_iu8(
                     true, A[i],  true, B[j],
                    acc[i][j],  false,  false);
    }

    const float s = alpha[0] * (1.0f / 127.0f);
    const int col = lane & 15;
    const int hi = (lane >> 4) << 3;
#pragma unroll
    for (int i = 0; i < 4; ++i) {
        const int n0 = (ntBase + i) * 16 + hi;
        float g[8];
#pragma unroll
        for (int j = 0; j < 8; ++j) g[j] = gamma[n0 + j] * s;
#pragma unroll
        for (int jt = 0; jt < 4; ++jt) {
            const int m = (mtBase + jt) * 16 + col;
#pragma unroll
            for (int j = 0; j < 8; ++j)
                out[(size_t)(n0 + j) * FAN_OUT + m] = (float)acc[i][jt][j] * g[j];
        }
    }
}

extern "C" void kernel_launch(void* const* d_in, const int* in_sizes, int n_in,
                              void* d_out, int out_size, void* d_ws, size_t ws_size,
                              hipStream_t stream) {
    const float* x = (const float*)d_in[0];
    const float* w = (const float*)d_in[1];
    float* out = (float*)d_out;
    char* ws = (char*)d_ws;

    float* part        = (float*)(ws + OFF_PART);
    float* alpha       = (float*)(ws + OFF_ALPHA);
    float* gamma       = (float*)(ws + OFF_GAMMA);
    signed char* qx    = (signed char*)(ws + OFF_QX);
    signed char* qw    = (signed char*)(ws + OFF_QW);

    absmean_partial<<<NPART, 256, 0, stream>>>(w, part);
    absmean_final<<<1, 256, 0, stream>>>(part, alpha);
    rowmax<<<NTOK, 256, 0, stream>>>(x, gamma);
    quant_x<<<(NT_TILES * KTILES * 32) / 256, 256, 0, stream>>>(x, gamma, qx);
    quant_w<<<(MT_TILES * KTILES * 32) / 256, 256, 0, stream>>>(w, alpha, qw);

    dim3 grid(FAN_OUT / 256, NTOK / 128);
    gemm_iu8<<<grid, 256, 0, stream>>>(qx, qw, gamma, alpha, out);
}
